// RelativePositionalEncoding_19121194402079
// MI455X (gfx1250) — matrix operand order repaired and checked by reading, not compiled, not run on hardware
//
#include <hip/hip_runtime.h>

// CDNA5 / gfx1250 WMMA kernel for relative positional encoding MLP.
// Per tile of 16 edges:  D(16x16) = GELU(rel*W1+b1)(16x64) x W2(64x16) + b2
// Layer 2: two chained v_wmma_f32_16x16x32_f16.
// Layer 1: v_pk_fma_f32 pre-activation + packed-f16 GELU (v_pk_*_f16 /
//          v_exp_f16 / v_rcp_f16), producing the WMMA A-fragment in place.

typedef __attribute__((ext_vector_type(16))) _Float16 v16h;
typedef __attribute__((ext_vector_type(8)))  float    v8f;
typedef __attribute__((ext_vector_type(2)))  _Float16 h2;
typedef __attribute__((ext_vector_type(2)))  float    f2;

#define KNBR    32
#define NHEADS  16
#define HIDDEN  64

// pack two f32 -> half2 (v_cvt_pk_f16_f32)
__device__ __forceinline__ h2 cvt_pk_h2(float a, float b) {
#if __has_builtin(__builtin_amdgcn_cvt_pkrtz)
  return __builtin_bit_cast(h2, __builtin_amdgcn_cvt_pkrtz(a, b));
#else
  h2 r = {(_Float16)a, (_Float16)b};
  return r;
#endif
}

// GELU (tanh form) on a packed half2:
//   gelu(x) = x / (1 + exp2(x*(A + B*x^2))),  A = -2.3022142, B = -0.1029484
// Limits are handled by f16 inf/zero: x<<0 -> exp2 -> +inf -> rcp -> 0 -> y=0.
__device__ __forceinline__ h2 gelu_h2(h2 x) {
  const h2 A = (_Float16)(-2.3022142f);
  const h2 B = (_Float16)(-0.1029484f);
  h2 x2  = x * x;                                    // v_pk_mul_f16
  h2 q   = __builtin_elementwise_fma(x2, B, A);      // v_pk_fma_f16
  h2 arg = x * q;                                    // v_pk_mul_f16
#if __has_builtin(__builtin_exp2f16)
  _Float16 e0 = __builtin_exp2f16(arg.x);            // v_exp_f16
  _Float16 e1 = __builtin_exp2f16(arg.y);
#else
  _Float16 e0 = (_Float16)__builtin_amdgcn_exp2f((float)arg.x);
  _Float16 e1 = (_Float16)__builtin_amdgcn_exp2f((float)arg.y);
#endif
  h2 d = (h2){e0, e1} + (h2)((_Float16)1.0f);        // v_pk_add_f16
#if __has_builtin(__builtin_amdgcn_rcph)
  h2 r = (h2){__builtin_amdgcn_rcph(d.x),            // v_rcp_f16
              __builtin_amdgcn_rcph(d.y)};
#else
  h2 r = (h2){(_Float16)((_Float16)1.0f / d.x),
              (_Float16)((_Float16)1.0f / d.y)};
#endif
  return x * r;                                      // v_pk_mul_f16
}

__global__ __launch_bounds__(256)
void rpe_wmma_kernel(const float* __restrict__ pos,
                     const int*   __restrict__ idx,
                     const float* __restrict__ w1,   // (3, 64) row-major
                     const float* __restrict__ b1,   // (64)
                     const float* __restrict__ w2,   // (64, 16) row-major
                     const float* __restrict__ b2,   // (16)
                     float* __restrict__ out,        // (N, 16, 32)
                     int npts, int ntiles) {
  // LDS, pair-major: for hidden pair p (units 2p, 2p+1), 8 floats at p*8:
  //   {wx0, wx1, wy0, wy1, wz0, wz1, b0, b1}  -> two ds_load_b128 per pair.
  __shared__ __align__(16) float wlds[HIDDEN * 4];

  const int tid = threadIdx.x;
  if (tid < HIDDEN * 4) {
    int p = tid >> 3, c = tid & 7;
    int j = 2 * p + (c & 1), comp = c >> 1;
    wlds[tid] = (comp < 3) ? w1[comp * HIDDEN + j] : b1[j];
  }
  __syncthreads();
  const float4* wp = (const float4*)wlds;

  const int lane = tid & 31;
  const int head = lane & 15;   // B/C/D column (head) index = lane % 16
  const int hi   = lane >> 4;   // lane group

  // --- B fragments: W2 as 32x16 f16 B-operands (K=0..31 and K=32..63).
  // Dense-B layout (analogous to documented sparse B 64x16):
  //   lanes 0-15 hold K = {0..7,16..23}, lanes 16-31 K = {8..15,24..31} (matches A).
  v16h bf_lo, bf_hi;
  #pragma unroll
  for (int h = 0; h < 16; ++h) {
    int k = h + (h & 8) + (hi << 3);
    bf_lo[h] = (_Float16)w2[k * NHEADS + head];
    bf_hi[h] = (_Float16)w2[(k + 32) * NHEADS + head];
  }
  const float bias2 = b2[head];

  // Wave-uniform grid-stride over tiles (EXEC stays all-ones for WMMA).
  int wave  = __builtin_amdgcn_readfirstlane((int)(blockIdx.x * blockDim.x + tid) >> 5);
  int nwave = (int)(gridDim.x * blockDim.x) >> 5;

  for (int tile = wave; tile < ntiles; tile += nwave) {
    const int npt = tile >> 1;          // point index (uniform -> scalar loads)
    const int kb  = (tile & 1) << 4;    // neighbor base: 0 or 16
    const int m   = head;               // A-matrix row = edge within tile

    // Edge index: npt*32 + kb + m == tile*16 + m.
    const int e = tile * 16 + m;
    const int j = idx[e];
    __builtin_prefetch(idx + (size_t)(tile + nwave) * 16 + m, 0, 0); // global_prefetch_b8

    const float cx = pos[npt * 3 + 0];  // uniform: s_load path
    const float cy = pos[npt * 3 + 1];
    const float cz = pos[npt * 3 + 2];
    f2 rx2 = pos[j * 3 + 0] - cx;       // splat to packed f32
    f2 ry2 = pos[j * 3 + 1] - cy;
    f2 rz2 = pos[j * 3 + 2] - cz;

    // --- Layer 1 + GELU, computed directly in WMMA A-fragment layout.
    // 16-bit A 16x32 (ISA table): lane<16 row m holds K={0..7,16..23},
    // lanes>=16 hold K={8..15,24..31}:  j1 = h + (h&8) + (hi<<3).
    // Process half-slots in pairs (hh even -> hidden units j1, j1+1).
    v16h af_lo, af_hi;
    #pragma unroll
    for (int hh = 0; hh < 16; hh += 2) {
      const int p = (hh + (hh & 8) + (hi << 3)) >> 1;   // hidden pair (lo frag)

      // lo fragment: hidden pair p  (units 2p, 2p+1 in 0..31)
      {
        float4 q0 = wp[p * 2 + 0];                      // ds_load_b128
        float4 q1 = wp[p * 2 + 1];                      // ds_load_b128
        f2 pre = {q1.z, q1.w};                          // bias pair
        pre = __builtin_elementwise_fma(rz2, (f2){q1.x, q1.y}, pre); // v_pk_fma_f32
        pre = __builtin_elementwise_fma(ry2, (f2){q0.z, q0.w}, pre);
        pre = __builtin_elementwise_fma(rx2, (f2){q0.x, q0.y}, pre);
        h2 g = gelu_h2(cvt_pk_h2(pre.x, pre.y));
        af_lo[hh]     = g.x;
        af_lo[hh + 1] = g.y;
      }
      // hi fragment: hidden pair p+16 (units 2p+32, 2p+33 in 32..63)
      {
        float4 q0 = wp[(p + 16) * 2 + 0];
        float4 q1 = wp[(p + 16) * 2 + 1];
        f2 pre = {q1.z, q1.w};
        pre = __builtin_elementwise_fma(rz2, (f2){q1.x, q1.y}, pre);
        pre = __builtin_elementwise_fma(ry2, (f2){q0.z, q0.w}, pre);
        pre = __builtin_elementwise_fma(rx2, (f2){q0.x, q0.y}, pre);
        h2 g = gelu_h2(cvt_pk_h2(pre.x, pre.y));
        af_hi[hh]     = g.x;
        af_hi[hh + 1] = g.y;
      }
    }

    // --- Layer 2: two chained WMMAs, C preloaded with b2 broadcast.
    v8f c;
    #pragma unroll
    for (int r = 0; r < 8; ++r) c[r] = bias2;
    c = __builtin_amdgcn_wmma_f32_16x16x32_f16(false, af_lo, false, bf_lo,
                                               (short)0, c, false, false);
    c = __builtin_amdgcn_wmma_f32_16x16x32_f16(false, af_hi, false, bf_hi,
                                               (short)0, c, false, false);

    // --- Store. D element (m,n): lane holds n=head, rows m = r + hi*8.
    // m maps to neighbor slot kb+m -> 8 contiguous floats per lane.
    float* obase = out + (size_t)npt * (NHEADS * KNBR) + head * KNBR + kb + (hi << 3);
    float4 o0 = {c[0], c[1], c[2], c[3]};
    float4 o1 = {c[4], c[5], c[6], c[7]};
    *(float4*)(obase + 0) = o0;                   // global_store_b128
    *(float4*)(obase + 4) = o1;                   // global_store_b128
  }
}

extern "C" void kernel_launch(void* const* d_in, const int* in_sizes, int n_in,
                              void* d_out, int out_size, void* d_ws, size_t ws_size,
                              hipStream_t stream) {
  (void)n_in; (void)out_size; (void)d_ws; (void)ws_size;
  const float* pos = (const float*)d_in[0];
  const int*   idx = (const int*)d_in[1];
  const float* w1  = (const float*)d_in[2];
  const float* b1  = (const float*)d_in[3];
  const float* w2  = (const float*)d_in[4];
  const float* b2  = (const float*)d_in[5];
  float* out = (float*)d_out;

  const int npts   = in_sizes[0] / 3;          // pos is (N,3)
  const int ntiles = (npts * KNBR) / 16;       // 16 edges per WMMA tile

  const int blocks = 2048;                     // 16384 waves, ~12 tiles each
  rpe_wmma_kernel<<<blocks, 256, 0, stream>>>(pos, idx, w1, b1, w2, b2, out,
                                              npts, ntiles);
}
